// CausalSelfAttention_83270825935224
// MI455X (gfx1250) — matrix operand order repaired and checked by reading, not compiled, not run on hardware
//
#include <hip/hip_runtime.h>
#include <hip/hip_bf16.h>

// ---------------------------------------------------------------------------
// Causal self-attention for MI455X (gfx1250), wave32 + WMMA bf16.
//   B=4, T=2048, D=1024, H=16, Dh=64
// Pipeline: k_qkv (x@W^T -> Q*scale, K, V^T bf16 ws) ->
//           k_attn (flash attention, online softmax, async-LDS tile staging) ->
//           k_oproj (Y@Wo^T + b_o -> f32 out)
// ---------------------------------------------------------------------------

typedef __bf16 bf16;
typedef __attribute__((ext_vector_type(16))) __bf16 bf16x16;
typedef __attribute__((ext_vector_type(8)))  __bf16 bf16x8;
typedef __attribute__((ext_vector_type(8)))  float  f32x8;

#define LDW 72   // LDS leading dim (elems): 144B rows -> 16B aligned, conflict-free
#define ASYNC_COPY 1

static __device__ inline f32x8 f32x8_zero() {
    f32x8 z;
#pragma unroll
    for (int i = 0; i < 8; ++i) z[i] = 0.0f;
    return z;
}

static __device__ inline unsigned pack_bf16(float a, float b) {
    union { unsigned u; __bf16 h[2]; } p;
    p.h[0] = (__bf16)a; p.h[1] = (__bf16)b;
    return p.u;
}

static __device__ inline bf16x16 frag_cat(bf16x8 lo, bf16x8 hh) {
    bf16x16 r;
#pragma unroll
    for (int i = 0; i < 8; ++i) { r[i] = lo[i]; r[i + 8] = hh[i]; }
    return r;
}

// A fragment (16x32, 16-bit): lane L holds row M=L&15.
// elems 0..7  -> K = k0 + 8*(L>=16) + 0..7 ; elems 8..15 -> +16
static __device__ inline bf16x16 load_frag_a(const bf16* p, int ld, int row, int k0, int hi) {
    const bf16* q = p + row * ld + k0 + hi * 8;
    bf16x8 lo = *(const bf16x8*)(q);
    bf16x8 hh = *(const bf16x8*)(q + 16);
    return frag_cat(lo, hh);
}

// B fragment (32x16, 16-bit), source stored [n][k] row-major:
// lane L holds col N=L&15; elems 0..7 -> K = k0 + 8*(L>=16) + 0..7 ; elems 8..15 -> +16
static __device__ inline bf16x16 load_frag_b(const bf16* p, int ld, int nrow, int k0, int hi) {
    const bf16* q = p + nrow * ld + k0 + hi * 8;
    bf16x8 lo = *(const bf16x8*)(q);
    bf16x8 hh = *(const bf16x8*)(q + 16);
    return frag_cat(lo, hh);
}

static __device__ inline f32x8 wmma_bf16(bf16x16 a, bf16x16 b, f32x8 c) {
    return __builtin_amdgcn_wmma_f32_16x16x32_bf16(false, a, false, b, (short)0, c, false, false);
}

// ---------------------------------------------------------------------------
// Kernel 1: QKV projection. grid = (16 n-tiles, 64 m-tiles, 3 matrices).
// Block tile 128x64, 4 waves x (32x64) = 2x4 WMMA tiles per wave.
// Q,K -> [bh][t][d] bf16 (Q scaled by Dh^-0.5); V -> [bh][d][t]
// ---------------------------------------------------------------------------
__global__ __launch_bounds__(128) void k_qkv(
    const float* __restrict__ x,
    const float* __restrict__ Wq, const float* __restrict__ Wk,
    const float* __restrict__ Wv,
    bf16* __restrict__ Qw, bf16* __restrict__ Kw, bf16* __restrict__ Vtw)
{
    __shared__ __align__(16) bf16 As[128 * LDW];
    __shared__ __align__(16) bf16 Bs[64 * LDW];

    const int tid  = threadIdx.x;
    const int lane = tid & 31;
    const int wid  = tid >> 5;
    const int hi   = lane >> 4;
    const int l16  = lane & 15;
    const int nt   = blockIdx.x, mt = blockIdx.y, mode = blockIdx.z;
    const int m0 = mt * 128, n0 = nt * 64;
    const float* W = (mode == 0) ? Wq : (mode == 1) ? Wk : Wv;
    const int m_off = wid * 32;

    f32x8 acc[2][4];
#pragma unroll
    for (int i = 0; i < 2; ++i)
#pragma unroll
        for (int j = 0; j < 4; ++j) acc[i][j] = f32x8_zero();

    for (int k0 = 0; k0 < 1024; k0 += 32) {
        __syncthreads();
#pragma unroll
        for (int i = 0; i < 16; ++i) {            // A: 128x32 f32 -> 2048 float2
            int idx = i * 128 + tid;
            int r = idx >> 4, c2 = (idx & 15) * 2;
            float2 v = *(const float2*)&x[(size_t)(m0 + r) * 1024 + k0 + c2];
            *(unsigned*)&As[r * LDW + c2] = pack_bf16(v.x, v.y);
        }
#pragma unroll
        for (int i = 0; i < 8; ++i) {             // B: 64x32 f32 -> 1024 float2
            int idx = i * 128 + tid;
            int r = idx >> 4, c2 = (idx & 15) * 2;
            float2 v = *(const float2*)&W[(size_t)(n0 + r) * 1024 + k0 + c2];
            *(unsigned*)&Bs[r * LDW + c2] = pack_bf16(v.x, v.y);
        }
        __syncthreads();
        bf16x16 a0 = load_frag_a(As, LDW, m_off + l16,      0, hi);
        bf16x16 a1 = load_frag_a(As, LDW, m_off + 16 + l16, 0, hi);
#pragma unroll
        for (int j = 0; j < 4; ++j) {
            bf16x16 bj = load_frag_b(Bs, LDW, j * 16 + l16, 0, hi);
            acc[0][j] = wmma_bf16(a0, bj, acc[0][j]);
            acc[1][j] = wmma_bf16(a1, bj, acc[1][j]);
        }
    }

    // Epilogue: uniform mode branch hoisted out of the store loops.
    const float qs = 0.125f;                      // Dh^-0.5
    if (mode == 0) {
#pragma unroll
        for (int i = 0; i < 2; ++i)
#pragma unroll
            for (int j = 0; j < 4; ++j)
#pragma unroll
                for (int r = 0; r < 8; ++r) {
                    int grow = m0 + m_off + i * 16 + r + hi * 8;
                    int gcol = n0 + j * 16 + l16;
                    size_t bh = (size_t)((grow >> 11) * 16 + (gcol >> 6));
                    Qw[(bh * 2048 + (grow & 2047)) * 64 + (gcol & 63)] =
                        (bf16)(acc[i][j][r] * qs);
                }
    } else if (mode == 1) {
#pragma unroll
        for (int i = 0; i < 2; ++i)
#pragma unroll
            for (int j = 0; j < 4; ++j)
#pragma unroll
                for (int r = 0; r < 8; ++r) {
                    int grow = m0 + m_off + i * 16 + r + hi * 8;
                    int gcol = n0 + j * 16 + l16;
                    size_t bh = (size_t)((grow >> 11) * 16 + (gcol >> 6));
                    Kw[(bh * 2048 + (grow & 2047)) * 64 + (gcol & 63)] =
                        (bf16)acc[i][j][r];
                }
    } else {
#pragma unroll
        for (int i = 0; i < 2; ++i)
#pragma unroll
            for (int j = 0; j < 4; ++j)
#pragma unroll
                for (int r = 0; r < 8; ++r) {
                    int grow = m0 + m_off + i * 16 + r + hi * 8;
                    int gcol = n0 + j * 16 + l16;
                    size_t bh = (size_t)((grow >> 11) * 16 + (gcol >> 6));
                    Vtw[(bh * 64 + (gcol & 63)) * 2048 + (grow & 2047)] =
                        (bf16)acc[i][j][r];
                }
    }
}

// ---------------------------------------------------------------------------
// Kernel 2: flash attention. grid.x = B*H*32 q-tiles; 4 waves x 16 q rows.
// K/V^T tiles staged in LDS via GLOBAL_LOAD_ASYNC_TO_LDS (ASYNCcnt).
// ---------------------------------------------------------------------------
__global__ __launch_bounds__(128) void k_attn(
    const bf16* __restrict__ Qw, const bf16* __restrict__ Kw,
    const bf16* __restrict__ Vtw, bf16* __restrict__ Yw)
{
    __shared__ __align__(16) bf16 Ks[64 * LDW];       // [key j][d]
    __shared__ __align__(16) bf16 Vs[64 * LDW];       // [d][key j]
    __shared__ __align__(16) bf16 Ps[4 * 16 * LDW];   // per-wave P staging

    const int tid  = threadIdx.x;
    const int lane = tid & 31;
    const int wid  = tid >> 5;
    const int hi   = lane >> 4;
    const int l16  = lane & 15;
    const int bh   = blockIdx.x >> 5;
    const int qt   = blockIdx.x & 31;
    const int b    = bh >> 4, h = bh & 15;
    const bf16* Qp = Qw  + (size_t)bh * 2048 * 64;
    const bf16* Kp = Kw  + (size_t)bh * 2048 * 64;
    const bf16* Vp = Vtw + (size_t)bh * 64 * 2048;
    const int qbase = qt * 64;
    const int qrow0 = qbase + wid * 16;

    bf16x16 qf0 = load_frag_a(Qp, 64, qrow0 + l16, 0,  hi);
    bf16x16 qf1 = load_frag_a(Qp, 64, qrow0 + l16, 32, hi);

    f32x8 oacc[4];
#pragma unroll
    for (int n = 0; n < 4; ++n) oacc[n] = f32x8_zero();
    float mrow[8], lrow[8];
#pragma unroll
    for (int r = 0; r < 8; ++r) { mrow[r] = -__builtin_inff(); lrow[r] = 0.0f; }

    for (int kt = 0; kt <= qt; ++kt) {
        const int kb = kt * 64;
        __syncthreads();
#pragma unroll
        for (int i = 0; i < 8; ++i) {             // 2 tiles x 4096 bf16, 8B per op
            int idx = i * 128 + tid;              // 0..1023
            int r  = idx >> 4;                    // 0..63
            int c4 = (idx & 15) * 4;              // 0..60
            const bf16* gk = Kp + (size_t)(kb + r) * 64 + c4;
            const bf16* gv = Vp + (size_t)r * 2048 + kb + c4;
#if ASYNC_COPY
            // Flat->LDS aperture: low 32 bits of the generic pointer are the
            // LDS byte address, which is what VDST must hold.
            unsigned kdst = (unsigned)(size_t)(&Ks[r * LDW + c4]);
            unsigned vdst = (unsigned)(size_t)(&Vs[r * LDW + c4]);
            asm volatile("global_load_async_to_lds_b64 %0, %1, off"
                         :: "v"(kdst), "v"(gk) : "memory");
            asm volatile("global_load_async_to_lds_b64 %0, %1, off"
                         :: "v"(vdst), "v"(gv) : "memory");
#else
            *(uint2*)&Ks[r * LDW + c4] = *(const uint2*)gk;
            *(uint2*)&Vs[r * LDW + c4] = *(const uint2*)gv;
#endif
        }
#if ASYNC_COPY
        asm volatile("s_wait_asynccnt 0x0" ::: "memory");
#endif
        __syncthreads();

        // S = Q K^T  (16x64 per wave, 4 j-subtiles)
        f32x8 st[4];
#pragma unroll
        for (int j = 0; j < 4; ++j) {
            f32x8 s = f32x8_zero();
            bf16x16 kb0 = load_frag_b(Ks, LDW, j * 16 + l16, 0,  hi);
            bf16x16 kb1 = load_frag_b(Ks, LDW, j * 16 + l16, 32, hi);
            s = wmma_bf16(qf0, kb0, s);
            s = wmma_bf16(qf1, kb1, s);
            st[j] = s;
        }

        if (kt == qt) {                           // causal mask, diagonal tile only
#pragma unroll
            for (int j = 0; j < 4; ++j)
#pragma unroll
                for (int r = 0; r < 8; ++r) {
                    int q = qrow0 + r + hi * 8;
                    int k = kb + j * 16 + l16;
                    if (k > q) st[j][r] = -__builtin_inff();
                }
        }

        // online softmax (row reductions within 16-lane halves)
#pragma unroll
        for (int r = 0; r < 8; ++r) {
            float mx = fmaxf(fmaxf(st[0][r], st[1][r]), fmaxf(st[2][r], st[3][r]));
            mx = fmaxf(mx, __shfl_xor(mx, 1));
            mx = fmaxf(mx, __shfl_xor(mx, 2));
            mx = fmaxf(mx, __shfl_xor(mx, 4));
            mx = fmaxf(mx, __shfl_xor(mx, 8));
            float mnew  = fmaxf(mrow[r], mx);
            float alpha = __expf(mrow[r] - mnew);
            float rs = 0.0f;
#pragma unroll
            for (int j = 0; j < 4; ++j) {
                float p = __expf(st[j][r] - mnew);
                st[j][r] = p;
                rs += p;
            }
            rs += __shfl_xor(rs, 1);
            rs += __shfl_xor(rs, 2);
            rs += __shfl_xor(rs, 4);
            rs += __shfl_xor(rs, 8);
            lrow[r] = lrow[r] * alpha + rs;
            mrow[r] = mnew;
#pragma unroll
            for (int n = 0; n < 4; ++n) oacc[n][r] *= alpha;
        }

        // P (C-layout f32) -> per-wave LDS region as bf16 A-source
        bf16* Pw = Ps + wid * 16 * LDW;
#pragma unroll
        for (int j = 0; j < 4; ++j)
#pragma unroll
            for (int r = 0; r < 8; ++r)
                Pw[(r + hi * 8) * LDW + j * 16 + l16] = (bf16)st[j][r];

        // O += P @ V
        bf16x16 pa0 = load_frag_a(Pw, LDW, l16, 0,  hi);
        bf16x16 pa1 = load_frag_a(Pw, LDW, l16, 32, hi);
#pragma unroll
        for (int n = 0; n < 4; ++n) {
            bf16x16 vb0 = load_frag_b(Vs, LDW, n * 16 + l16, 0,  hi);
            bf16x16 vb1 = load_frag_b(Vs, LDW, n * 16 + l16, 32, hi);
            oacc[n] = wmma_bf16(pa0, vb0, oacc[n]);
            oacc[n] = wmma_bf16(pa1, vb1, oacc[n]);
        }
    }

    // normalize + store Y as [b][t][h*64+d] bf16
#pragma unroll
    for (int r = 0; r < 8; ++r) {
        float inv = 1.0f / lrow[r];
        int t = qrow0 + r + hi * 8;
#pragma unroll
        for (int n = 0; n < 4; ++n) {
            int d = n * 16 + l16;
            Yw[((size_t)(b * 2048 + t)) * 1024 + h * 64 + d] = (bf16)(oacc[n][r] * inv);
        }
    }
}

// ---------------------------------------------------------------------------
// Kernel 3: output projection Y @ Wo^T + b_o -> f32. grid = (16, 64).
// Block tile 128x64, 4 waves x (32x64).
// ---------------------------------------------------------------------------
__global__ __launch_bounds__(128) void k_oproj(
    const bf16* __restrict__ Yw, const float* __restrict__ Wo,
    const float* __restrict__ bo, float* __restrict__ out)
{
    __shared__ __align__(16) bf16 As[128 * LDW];
    __shared__ __align__(16) bf16 Bs[64 * LDW];

    const int tid  = threadIdx.x;
    const int lane = tid & 31;
    const int wid  = tid >> 5;
    const int hi   = lane >> 4;
    const int l16  = lane & 15;
    const int nt = blockIdx.x, mt = blockIdx.y;
    const int m0 = mt * 128, n0 = nt * 64;
    const int m_off = wid * 32;

    f32x8 acc[2][4];
#pragma unroll
    for (int i = 0; i < 2; ++i)
#pragma unroll
        for (int j = 0; j < 4; ++j) acc[i][j] = f32x8_zero();

    for (int k0 = 0; k0 < 1024; k0 += 32) {
        __syncthreads();
#pragma unroll
        for (int i = 0; i < 8; ++i) {             // A: 128x32 bf16 = 1024 uint2
            int idx = i * 128 + tid;
            int r = idx >> 3, c4 = (idx & 7) * 4;
            *(uint2*)&As[r * LDW + c4] =
                *(const uint2*)(Yw + (size_t)(m0 + r) * 1024 + k0 + c4);
        }
#pragma unroll
        for (int i = 0; i < 8; ++i) {             // B: 64x32 f32 -> 1024 float2
            int idx = i * 128 + tid;
            int r = idx >> 4, c2 = (idx & 15) * 2;
            float2 v = *(const float2*)&Wo[(size_t)(n0 + r) * 1024 + k0 + c2];
            *(unsigned*)&Bs[r * LDW + c2] = pack_bf16(v.x, v.y);
        }
        __syncthreads();
        bf16x16 a0 = load_frag_a(As, LDW, m_off + l16,      0, hi);
        bf16x16 a1 = load_frag_a(As, LDW, m_off + 16 + l16, 0, hi);
#pragma unroll
        for (int j = 0; j < 4; ++j) {
            bf16x16 bj = load_frag_b(Bs, LDW, j * 16 + l16, 0, hi);
            acc[0][j] = wmma_bf16(a0, bj, acc[0][j]);
            acc[1][j] = wmma_bf16(a1, bj, acc[1][j]);
        }
    }

#pragma unroll
    for (int i = 0; i < 2; ++i)
#pragma unroll
        for (int j = 0; j < 4; ++j)
#pragma unroll
            for (int r = 0; r < 8; ++r) {
                int grow = m0 + m_off + i * 16 + r + hi * 8;
                int gcol = n0 + j * 16 + l16;
                out[(size_t)grow * 1024 + gcol] = acc[i][j][r] + bo[gcol];
            }
}

// ---------------------------------------------------------------------------
extern "C" void kernel_launch(void* const* d_in, const int* in_sizes, int n_in,
                              void* d_out, int out_size, void* d_ws, size_t ws_size,
                              hipStream_t stream)
{
    (void)in_sizes; (void)n_in; (void)out_size; (void)ws_size;
    const float* x  = (const float*)d_in[0];
    const float* Wq = (const float*)d_in[1];
    const float* Wk = (const float*)d_in[2];
    const float* Wv = (const float*)d_in[3];
    const float* Wo = (const float*)d_in[4];
    const float* bo = (const float*)d_in[5];
    float* out = (float*)d_out;

    const size_t ELEMS = (size_t)4 * 16 * 2048 * 64;   // 8,388,608 per tensor
    bf16* Qw  = (bf16*)d_ws;
    bf16* Kw  = Qw  + ELEMS;
    bf16* Vtw = Kw  + ELEMS;
    bf16* Yw  = Vtw + ELEMS;

    k_qkv  <<<dim3(16, 64, 3), 128, 0, stream>>>(x, Wq, Wk, Wv, Qw, Kw, Vtw);
    k_attn <<<dim3(64 * 32),   128, 0, stream>>>(Qw, Kw, Vtw, Yw);
    k_oproj<<<dim3(16, 64),    128, 0, stream>>>(Yw, Wo, bo, out);
}
